// ConsistencyLoss_11244224381575
// MI455X (gfx1250) — matrix operand order repaired and checked by reading, not compiled, not run on hardware
//
#include <hip/hip_runtime.h>

// ---------------------------------------------------------------------------
// Problem constants (fixed by setup_inputs): B=8, C=256, H=W=48, N=2304
// ---------------------------------------------------------------------------
#define BB 8
#define CC 256
#define NN 2304
#define DIST_THR 31.5f
#define EPS 1e-8f

typedef __attribute__((ext_vector_type(16))) __bf16 v16bf;
typedef __attribute__((ext_vector_type(8)))  __bf16 v8bf;
typedef __attribute__((ext_vector_type(8)))  float  v8f;

// ws layout (bytes):
//   4 bf16 tensors [B][N][C], order: A1=y, B1=zp, A2=z, B2=yp
#define TSZ ((size_t)BB * NN * CC)          // elements per bf16 tensor
#define WS_BF16_BYTES (4 * TSZ * 2)
//   4 inv-norm arrays [B*N] f32, same order
#define RNSZ ((size_t)BB * NN)
#define WS_RN_OFF WS_BF16_BYTES
#define WS_RN_BYTES (4 * RNSZ * 4)
//   per-block partials
#define GEMM_GX (NN / 32)                   // 72 m-supertiles (2 m-tiles / wave)
#define GEMM_GY (NN / (32 * 8))             // 9 n-strips (8 waves x 32 rows)
#define GEMM_NBLK (GEMM_GX * GEMM_GY * BB)  // 5184
#define WS_SUM_OFF (WS_RN_OFF + WS_RN_BYTES)
#define WS_CNT_OFF (WS_SUM_OFF + (size_t)GEMM_NBLK * 4)
#define WS_NEED    (WS_CNT_OFF + (size_t)GEMM_NBLK * 4)

// ---------------------------------------------------------------------------
// Kernel 1: transpose [B,C,N] f32 -> [B,N,C] bf16 for all four tensors
// grid (N/32, C/32, 4*B), block (32, 8)
// ---------------------------------------------------------------------------
__global__ __launch_bounds__(256) void k_transpose_bf16(
    const float* __restrict__ a1, const float* __restrict__ b1,
    const float* __restrict__ a2, const float* __restrict__ b2,
    __bf16* __restrict__ outbase) {
  __shared__ float tile[32][33];
  const int t = blockIdx.z & 3;
  const int b = blockIdx.z >> 2;
  const float* x = (t == 0) ? a1 : (t == 1) ? b1 : (t == 2) ? a2 : b2;
  __bf16* o = outbase + (size_t)t * TSZ + (size_t)b * NN * CC;
  const float* xb = x + (size_t)b * CC * NN;

  const int nt = blockIdx.x * 32;
  const int ct = blockIdx.y * 32;
  const int tx = threadIdx.x, ty = threadIdx.y;

#pragma unroll
  for (int i = 0; i < 4; ++i) {
    int c = ct + ty + i * 8;
    tile[ty + i * 8][tx] = xb[(size_t)c * NN + nt + tx];   // coalesced over n
  }
  __syncthreads();
#pragma unroll
  for (int i = 0; i < 4; ++i) {
    int n = nt + ty + i * 8;
    o[(size_t)n * CC + ct + tx] = (__bf16)tile[tx][ty + i * 8];  // coalesced over c
  }
}

// ---------------------------------------------------------------------------
// Kernel 2: inverse norms over channel dim, f32 source.
// grid (N/256, B, 4), block 256
// ---------------------------------------------------------------------------
__global__ __launch_bounds__(256) void k_inv_norms(
    const float* __restrict__ a1, const float* __restrict__ b1,
    const float* __restrict__ a2, const float* __restrict__ b2,
    float* __restrict__ rnbase) {
  const int t = blockIdx.z;
  const int b = blockIdx.y;
  const int n = blockIdx.x * blockDim.x + threadIdx.x;
  const float* x = (t == 0) ? a1 : (t == 1) ? b1 : (t == 2) ? a2 : b2;
  const float* col = x + (size_t)b * CC * NN + n;
  float s = 0.f;
#pragma unroll 8
  for (int c = 0; c < CC; ++c) {
    float v = col[(size_t)c * NN];   // consecutive threads -> consecutive n: coalesced
    s += v * v;
  }
  rnbase[(size_t)t * RNSZ + (size_t)b * NN + n] = 1.0f / fmaxf(sqrtf(s), EPS);
}

// ---------------------------------------------------------------------------
// WMMA fragment loaders (bf16 stored [B][N][C], row = n or m, contiguous in c)
// A 16x32 bf16: lane<16 -> M=lane,    K = {8h+0..7, 16+8h+0..7}, h = lane/16
// B 32x16 bf16: lane<16 -> N=lane%16, K = {8h+0..7, 16+8h+0..7}
// ---------------------------------------------------------------------------
__device__ __forceinline__ v16bf load_a_frag(const __bf16* row_k0, int half) {
  const v8bf lo = *(const v8bf*)(row_k0 + 8 * half);
  const v8bf hi = *(const v8bf*)(row_k0 + 16 + 8 * half);
  v16bf a;
#pragma unroll
  for (int i = 0; i < 8; ++i) { a[i] = lo[i]; a[i + 8] = hi[i]; }
  return a;
}
__device__ __forceinline__ v16bf load_b_frag(const __bf16* row_k0, int half) {
  const v8bf lo = *(const v8bf*)(row_k0 + 8 * half);
  const v8bf hi = *(const v8bf*)(row_k0 + 16 + 8 * half);
  v16bf r;
#pragma unroll
  for (int i = 0; i < 8; ++i) { r[i] = lo[i]; r[i + 8] = hi[i]; }
  return r;
}

// ---------------------------------------------------------------------------
// Kernel 3: masked batched dual-GEMM via v_wmma_f32_16x16x32_bf16.
// grid (72, 9, 8), block 256 = 8 waves.
// Wave w owns a 32x32 output macro-tile (2 n-tiles x 2 m-tiles) for BOTH
// matrix pairs -> each fragment load feeds two WMMAs (2 b128 loads / WMMA).
// ---------------------------------------------------------------------------
__global__ __launch_bounds__(256) void k_gemm_masked(
    const __bf16* __restrict__ tbase, const float* __restrict__ rnbase,
    const float* __restrict__ dist,
    float* __restrict__ blockSum, int* __restrict__ blockCnt) {
  const __bf16* tA1 = tbase;
  const __bf16* tB1 = tbase + 1 * TSZ;
  const __bf16* tA2 = tbase + 2 * TSZ;
  const __bf16* tB2 = tbase + 3 * TSZ;
  const float* rnA1 = rnbase;
  const float* rnB1 = rnbase + 1 * RNSZ;
  const float* rnA2 = rnbase + 2 * RNSZ;
  const float* rnB2 = rnbase + 3 * RNSZ;

  const int tid = threadIdx.x;
  const int wave = tid >> 5, lane = tid & 31;
  const int l = lane & 15, half = lane >> 4;
  const int b = blockIdx.z;
  const int m0 = blockIdx.x * 32;                 // 2 m-tiles: m0, m0+16
  const int n0 = blockIdx.y * 256 + wave * 32;    // 2 n-tiles: n0, n0+16

  const size_t rowA = ((size_t)b * NN + n0 + l) * CC;   // n-tile 0; +16*CC for tile 1
  const size_t rowB = ((size_t)b * NN + m0 + l) * CC;   // m-tile 0; +16*CC for tile 1
  const __bf16* pa1 = tA1 + rowA;
  const __bf16* pa2 = tA2 + rowA;
  const __bf16* pb1 = tB1 + rowB;
  const __bf16* pb2 = tB2 + rowB;

  v8f acc1[2][2] = {};  // y . zp   [n-tile][m-tile]
  v8f acc2[2][2] = {};  // z . yp
#pragma unroll
  for (int k0 = 0; k0 < CC; k0 += 32) {
    // pair 1: y x zp
    {
      v16bf a0 = load_a_frag(pa1 + k0, half);
      v16bf a1 = load_a_frag(pa1 + 16 * CC + k0, half);
      v16bf b0 = load_b_frag(pb1 + k0, half);
      v16bf b1 = load_b_frag(pb1 + 16 * CC + k0, half);
      acc1[0][0] = __builtin_amdgcn_wmma_f32_16x16x32_bf16(false, a0, false, b0,
                                                           (short)0, acc1[0][0], false, false);
      acc1[0][1] = __builtin_amdgcn_wmma_f32_16x16x32_bf16(false, a0, false, b1,
                                                           (short)0, acc1[0][1], false, false);
      acc1[1][0] = __builtin_amdgcn_wmma_f32_16x16x32_bf16(false, a1, false, b0,
                                                           (short)0, acc1[1][0], false, false);
      acc1[1][1] = __builtin_amdgcn_wmma_f32_16x16x32_bf16(false, a1, false, b1,
                                                           (short)0, acc1[1][1], false, false);
    }
    // pair 2: z x yp
    {
      v16bf a0 = load_a_frag(pa2 + k0, half);
      v16bf a1 = load_a_frag(pa2 + 16 * CC + k0, half);
      v16bf b0 = load_b_frag(pb2 + k0, half);
      v16bf b1 = load_b_frag(pb2 + 16 * CC + k0, half);
      acc2[0][0] = __builtin_amdgcn_wmma_f32_16x16x32_bf16(false, a0, false, b0,
                                                           (short)0, acc2[0][0], false, false);
      acc2[0][1] = __builtin_amdgcn_wmma_f32_16x16x32_bf16(false, a0, false, b1,
                                                           (short)0, acc2[0][1], false, false);
      acc2[1][0] = __builtin_amdgcn_wmma_f32_16x16x32_bf16(false, a1, false, b0,
                                                           (short)0, acc2[1][0], false, false);
      acc2[1][1] = __builtin_amdgcn_wmma_f32_16x16x32_bf16(false, a1, false, b1,
                                                           (short)0, acc2[1][1], false, false);
    }
  }

  // C/D layout: element j of acc[ti][tj] <-> (row = n0 + 16*ti + 8*half + j,
  //                                           col = m0 + 16*tj + (lane&15))
  float psum = 0.f;
  int pcnt = 0;
#pragma unroll
  for (int ti = 0; ti < 2; ++ti) {
    const int nb = n0 + 16 * ti + 8 * half;
#pragma unroll
    for (int tj = 0; tj < 2; ++tj) {
      const int m = m0 + 16 * tj + l;
      const float rb1 = rnB1[(size_t)b * NN + m];
      const float rb2 = rnB2[(size_t)b * NN + m];
#pragma unroll
      for (int j = 0; j < 8; ++j) {
        const int n = nb + j;
        const float d = dist[(size_t)n * NN + m];
        if (d < DIST_THR) {
          psum += acc1[ti][tj][j] * (rnA1[(size_t)b * NN + n] * rb1) +
                  acc2[ti][tj][j] * (rnA2[(size_t)b * NN + n] * rb2);
          ++pcnt;
        }
      }
    }
  }

  // wave32 butterfly reduction
#pragma unroll
  for (int off = 16; off > 0; off >>= 1) {
    psum += __shfl_xor(psum, off, 32);
    pcnt += __shfl_xor(pcnt, off, 32);
  }
  __shared__ float sS[8];
  __shared__ int sC[8];
  if (lane == 0) { sS[wave] = psum; sC[wave] = pcnt; }
  __syncthreads();
  if (tid == 0) {
    float ts = 0.f;
    int tc = 0;
#pragma unroll
    for (int w = 0; w < 8; ++w) { ts += sS[w]; tc += sC[w]; }
    const int bid = (blockIdx.z * gridDim.y + blockIdx.y) * gridDim.x + blockIdx.x;
    blockSum[bid] = ts;
    blockCnt[bid] = tc;
  }
}

// ---------------------------------------------------------------------------
// Kernel 4: deterministic final reduction (fixed strided order + fixed tree)
// ---------------------------------------------------------------------------
__global__ __launch_bounds__(256) void k_final_reduce(
    const float* __restrict__ blockSum, const int* __restrict__ blockCnt,
    int nblocks, float* __restrict__ out) {
  __shared__ float sS[256];
  __shared__ int sC[256];
  float s = 0.f;
  int c = 0;
  for (int i = threadIdx.x; i < nblocks; i += 256) {
    s += blockSum[i];
    c += blockCnt[i];
  }
  sS[threadIdx.x] = s;
  sC[threadIdx.x] = c;
  __syncthreads();
  for (int off = 128; off > 0; off >>= 1) {
    if (threadIdx.x < off) {
      sS[threadIdx.x] += sS[threadIdx.x + off];
      sC[threadIdx.x] += sC[threadIdx.x + off];
    }
    __syncthreads();
  }
  if (threadIdx.x == 0) out[0] = -sS[0] / (float)sC[0];
}

// ---------------------------------------------------------------------------
// launch: inputs in setup order: y, yp, z, zp, distances
// A1=y, B1=zp, A2=z, B2=yp
// ---------------------------------------------------------------------------
extern "C" void kernel_launch(void* const* d_in, const int* in_sizes, int n_in,
                              void* d_out, int out_size, void* d_ws, size_t ws_size,
                              hipStream_t stream) {
  (void)in_sizes; (void)n_in; (void)out_size;
  if (ws_size < WS_NEED) return;  // not enough scratch; bail deterministically

  const float* y  = (const float*)d_in[0];
  const float* yp = (const float*)d_in[1];
  const float* z  = (const float*)d_in[2];
  const float* zp = (const float*)d_in[3];
  const float* dist = (const float*)d_in[4];

  char* ws = (char*)d_ws;
  __bf16* tbase   = (__bf16*)ws;
  float*  rnbase  = (float*)(ws + WS_RN_OFF);
  float*  bSum    = (float*)(ws + WS_SUM_OFF);
  int*    bCnt    = (int*)(ws + WS_CNT_OFF);
  float*  out     = (float*)d_out;

  // 1) transpose + bf16 convert: grid (72, 8, 32), block (32, 8)
  k_transpose_bf16<<<dim3(NN / 32, CC / 32, 4 * BB), dim3(32, 8), 0, stream>>>(
      y, zp, z, yp, tbase);

  // 2) inverse norms (f32 source): grid (9, 8, 4), block 256
  k_inv_norms<<<dim3(NN / 256, BB, 4), dim3(256), 0, stream>>>(
      y, zp, z, yp, rnbase);

  // 3) masked dual-GEMM with WMMA (2x2 tiles/wave): grid (72, 9, 8), block 256
  k_gemm_masked<<<dim3(GEMM_GX, GEMM_GY, BB), dim3(256), 0, stream>>>(
      tbase, rnbase, dist, bSum, bCnt);

  // 4) final deterministic reduction
  k_final_reduce<<<dim3(1), dim3(256), 0, stream>>>(bSum, bCnt, GEMM_NBLK, out);
}
